// BitLinear_79293686219429
// MI455X (gfx1250) — compile-verified, matrix operand order checked by reading, not run on hardware
//
#include <hip/hip_runtime.h>
#include <stdint.h>

typedef __attribute__((ext_vector_type(8))) int v8i;

#define N_ROWS 8192
#define D_IN   4096
#define D_OUT  4096

#define BM 128
#define BN 128
#define BK 64
#define LDK 80   // padded LDS row stride (16B-aligned, conflict-free for b64 frag loads)

// ---- CDNA5 async global->LDS copy (guarded: falls back to VGPR staging) ----
#if defined(__HIP_DEVICE_COMPILE__) && __has_builtin(__builtin_amdgcn_global_load_async_to_lds_b128)
#define USE_ASYNC 1
#else
#define USE_ASYNC 0
#endif

// builtin parameter types: int4 pointers in global (AS1) / shared (AS3) address spaces
typedef __attribute__((vector_size(16))) int v4i;
typedef __attribute__((address_space(1))) v4i g_v4i;
typedef __attribute__((address_space(3))) v4i l_v4i;

static __device__ __forceinline__ g_v4i* gcast(const void* p) {
    return (g_v4i*)(unsigned long long)(uintptr_t)p;   // generic -> global (same 64-bit addr)
}
static __device__ __forceinline__ l_v4i* lcast(const void* p) {
    return (l_v4i*)(unsigned int)(uintptr_t)p;         // generic LDS addr[31:0] == LDS offset
}

// ---------------- pass 0: init scale slot ----------------
__global__ void init_kernel(unsigned* __restrict__ maxbits) { *maxbits = 0u; }

// ---------------- pass 1: global absmax over x ----------------
__global__ void absmax_kernel(const float* __restrict__ x,
                              unsigned* __restrict__ maxbits, int n4)
{
    const float4* x4 = (const float4*)x;
    float m = 0.0f;
    int stride = gridDim.x * blockDim.x;
    for (int i = blockIdx.x * blockDim.x + threadIdx.x; i < n4; i += stride) {
        float4 v = x4[i];
        m = fmaxf(m, fabsf(v.x));
        m = fmaxf(m, fabsf(v.y));
        m = fmaxf(m, fabsf(v.z));
        m = fmaxf(m, fabsf(v.w));
    }
    __shared__ float red[256];
    red[threadIdx.x] = m;
    __syncthreads();
    for (int s = 128; s > 0; s >>= 1) {
        if (threadIdx.x < s) red[threadIdx.x] = fmaxf(red[threadIdx.x], red[threadIdx.x + s]);
        __syncthreads();
    }
    // non-negative IEEE floats compare correctly as uints
    if (threadIdx.x == 0) atomicMax(maxbits, __float_as_uint(red[0]));
}

// ---------------- pass 2: quantize x -> int8 (packed 4/dword) ----------------
__global__ void quant_kernel(const float* __restrict__ x,
                             const unsigned* __restrict__ maxbits,
                             unsigned* __restrict__ xq, int n4)
{
    int i = blockIdx.x * blockDim.x + threadIdx.x;
    if (i >= n4) return;
    float mx = __uint_as_float(*maxbits);
    float s = mx * (1.0f / 127.0f);
    if (s == 0.0f) s = 1.0f;
    float inv = 1.0f / s;
    float4 v = ((const float4*)x)[i];
    int q0 = max(-128, min(127, (int)rintf(v.x * inv)));   // rintf = RNE, matches jnp.round
    int q1 = max(-128, min(127, (int)rintf(v.y * inv)));
    int q2 = max(-128, min(127, (int)rintf(v.z * inv)));
    int q3 = max(-128, min(127, (int)rintf(v.w * inv)));
    xq[i] = (unsigned)(q0 & 0xff) | ((unsigned)(q1 & 0xff) << 8) |
            ((unsigned)(q2 & 0xff) << 16) | ((unsigned)(q3 & 0xff) << 24);
}

// ---------------- pass 3: pack ternary int32 weights -> int8 ----------------
__global__ void packw_kernel(const int* __restrict__ w,
                             unsigned* __restrict__ wq, int n4)
{
    int i = blockIdx.x * blockDim.x + threadIdx.x;
    if (i >= n4) return;
    int4 v = ((const int4*)w)[i];
    wq[i] = (unsigned)(v.x & 0xff) | ((unsigned)(v.y & 0xff) << 8) |
            ((unsigned)(v.z & 0xff) << 16) | ((unsigned)(v.w & 0xff) << 24);
}

// ---------------- pass 4: int8 WMMA GEMM + scale + bias ----------------
// ISA 8-bit A/B fragment layout (16x64, MxK): lane L holds row (L%16);
// chunk i (2 dwords) at K byte-offset i*16 + (L/16)*8.
static __device__ __forceinline__ v8i load_frag(const unsigned char* p)
{
    uint2 c0 = *(const uint2*)(p +  0);
    uint2 c1 = *(const uint2*)(p + 16);
    uint2 c2 = *(const uint2*)(p + 32);
    uint2 c3 = *(const uint2*)(p + 48);
    v8i r;
    r[0] = (int)c0.x; r[1] = (int)c0.y;
    r[2] = (int)c1.x; r[3] = (int)c1.y;
    r[4] = (int)c2.x; r[5] = (int)c2.y;
    r[6] = (int)c3.x; r[7] = (int)c3.y;
    return r;
}

__global__ __launch_bounds__(256)
void gemm_kernel(const unsigned char* __restrict__ xq,
                 const unsigned char* __restrict__ wq,
                 const unsigned* __restrict__ maxbits,
                 const float* __restrict__ wscale,
                 const float* __restrict__ bias,
                 float* __restrict__ out)
{
    __shared__ __align__(16) unsigned char sm[2][2][BM][LDK]; // [buf][A/B][row][K] = 40 KB

    const int tid  = threadIdx.x;
    const int lane = tid & 31;
    const int w    = tid >> 5;     // wave 0..7
    const int wm   = w >> 2;       // 0..1 : 64-row band
    const int wn   = w & 3;        // 0..3 : 32-col band
    const int bM   = blockIdx.y * BM;
    const int bN   = blockIdx.x * BN;

    // staging: 256 threads x 16B cover 64 rows x 64B; two slices -> 128 rows
    const int srow = tid >> 2;           // 0..63
    const int scg  = (tid & 3) * 16;     // 0,16,32,48

    const unsigned char* aG = xq + (size_t)(bM + srow) * D_IN + scg;
    const unsigned char* bG = wq + (size_t)(bN + srow) * D_IN + scg;

    v8i acc[4][2];
    #pragma unroll
    for (int i = 0; i < 4; ++i)
        #pragma unroll
        for (int j = 0; j < 2; ++j)
            acc[i][j] = (v8i){};

#if !USE_ASYNC
    uint4 ra0, ra1, rb0, rb1;
#endif

    // Issue the tile copy for K-step `kt` into LDS buffer `dstbuf`.
    auto issue_next = [&](int dstbuf, int kt) {
        const size_t ko = (size_t)kt * BK;
#if USE_ASYNC
        // direct global->LDS DMA, tracked by ASYNCcnt; no VGPR staging
        __builtin_amdgcn_global_load_async_to_lds_b128(
            gcast(aG + ko),                      lcast(&sm[dstbuf][0][srow     ][scg]), 0, 0);
        __builtin_amdgcn_global_load_async_to_lds_b128(
            gcast(aG + ko + (size_t)64 * D_IN),  lcast(&sm[dstbuf][0][srow + 64][scg]), 0, 0);
        __builtin_amdgcn_global_load_async_to_lds_b128(
            gcast(bG + ko),                      lcast(&sm[dstbuf][1][srow     ][scg]), 0, 0);
        __builtin_amdgcn_global_load_async_to_lds_b128(
            gcast(bG + ko + (size_t)64 * D_IN),  lcast(&sm[dstbuf][1][srow + 64][scg]), 0, 0);
#else
        ra0 = *(const uint4*)(aG + ko);
        ra1 = *(const uint4*)(aG + ko + (size_t)64 * D_IN);
        rb0 = *(const uint4*)(bG + ko);
        rb1 = *(const uint4*)(bG + ko + (size_t)64 * D_IN);
#endif
    };
    // Make the copy visible in LDS (before the workgroup barrier).
    auto finish_next = [&](int dstbuf) {
#if USE_ASYNC
        (void)dstbuf;
#if __has_builtin(__builtin_amdgcn_s_wait_asynccnt)
        __builtin_amdgcn_s_wait_asynccnt(0);
#else
        asm volatile("s_wait_asynccnt 0x0" ::: "memory");
#endif
#else
        *(uint4*)&sm[dstbuf][0][srow     ][scg] = ra0;
        *(uint4*)&sm[dstbuf][0][srow + 64][scg] = ra1;
        *(uint4*)&sm[dstbuf][1][srow     ][scg] = rb0;
        *(uint4*)&sm[dstbuf][1][srow + 64][scg] = rb1;
#endif
    };

    const int NK = D_IN / BK;  // 64
    issue_next(0, 0);
    finish_next(0);
    __syncthreads();

    const int lg = lane >> 4;      // half-wave group
    const int lr = lane & 15;      // row/col within tile

    int buf = 0;
    for (int kt = 0; kt < NK; ++kt) {
        if (kt + 1 < NK) issue_next(buf ^ 1, kt + 1);
        if (kt + 2 < NK) {
            const size_t ko = (size_t)(kt + 2) * BK;
            __builtin_prefetch(aG + ko, 0, 1);
            __builtin_prefetch(aG + ko + (size_t)64 * D_IN, 0, 1);
            __builtin_prefetch(bG + ko, 0, 1);
            __builtin_prefetch(bG + ko + (size_t)64 * D_IN, 0, 1);
        }

        v8i af[4], bf[2];
        #pragma unroll
        for (int tm = 0; tm < 4; ++tm)
            af[tm] = load_frag(&sm[buf][0][wm * 64 + tm * 16 + lr][lg * 8]);
        #pragma unroll
        for (int tn = 0; tn < 2; ++tn)
            bf[tn] = load_frag(&sm[buf][1][wn * 32 + tn * 16 + lr][lg * 8]);

        #pragma unroll
        for (int tm = 0; tm < 4; ++tm)
            #pragma unroll
            for (int tn = 0; tn < 2; ++tn)
                acc[tm][tn] = __builtin_amdgcn_wmma_i32_16x16x64_iu8(
                    /*sgn_a=*/true, af[tm], /*sgn_b=*/true, bf[tn],
                    acc[tm][tn], /*reuse_a=*/false, /*reuse_b=*/false);

        if (kt + 1 < NK) finish_next(buf ^ 1);
        __syncthreads();
        buf ^= 1;
    }

    // epilogue: out = acc * (wscale * act_scale) + bias
    float mx = __uint_as_float(*maxbits);
    float s = mx * (1.0f / 127.0f);
    if (s == 0.0f) s = 1.0f;
    const float f = wscale[0] * s;

    #pragma unroll
    for (int tm = 0; tm < 4; ++tm) {
        const int row0 = bM + wm * 64 + tm * 16 + lg * 8;  // C/D: M = r + 8*(lane/16)
        #pragma unroll
        for (int tn = 0; tn < 2; ++tn) {
            const int col = bN + wn * 32 + tn * 16 + lr;   // N = lane%16
            const float bv = bias[col];
            #pragma unroll
            for (int r = 0; r < 8; ++r)
                out[(size_t)(row0 + r) * D_OUT + col] = (float)acc[tm][tn][r] * f + bv;
        }
    }
}

extern "C" void kernel_launch(void* const* d_in, const int* in_sizes, int n_in,
                              void* d_out, int out_size, void* d_ws, size_t ws_size,
                              hipStream_t stream)
{
    const float* x      = (const float*)d_in[0];
    const int*   wgt    = (const int*)d_in[1];
    const float* wscale = (const float*)d_in[2];
    const float* bias   = (const float*)d_in[3];
    float* out = (float*)d_out;

    unsigned char* ws = (unsigned char*)d_ws;
    unsigned* maxbits   = (unsigned*)ws;
    unsigned char* xq   = ws + 256;
    unsigned char* wq   = xq + (size_t)N_ROWS * D_IN;   // 32 MiB + 16 MiB workspace

    const int n4x = N_ROWS * D_IN / 4;   // 8,388,608
    const int n4w = D_OUT * D_IN / 4;    // 4,194,304

    init_kernel<<<1, 1, 0, stream>>>(maxbits);
    absmax_kernel<<<2048, 256, 0, stream>>>(x, maxbits, n4x);
    quant_kernel<<<(n4x + 255) / 256, 256, 0, stream>>>(x, maxbits, (unsigned*)xq, n4x);
    packw_kernel<<<(n4w + 255) / 256, 256, 0, stream>>>(wgt, (unsigned*)wq, n4w);

    dim3 grid(D_OUT / BN, N_ROWS / BM);  // 32 x 64
    gemm_kernel<<<grid, 256, 0, stream>>>(xq, wq, maxbits, wscale, bias, out);
}
